// TriangleMultiplicativeUpdate_66520453481108
// MI455X (gfx1250) — matrix operand order repaired and checked by reading, not compiled, not run on hardware
//
#include <hip/hip_runtime.h>
#include <hip/hip_fp16.h>

#define DEV_INLINE __device__ __forceinline__

typedef __attribute__((ext_vector_type(16))) _Float16 v16h;
typedef __attribute__((ext_vector_type(8)))  _Float16 v8h;
typedef __attribute__((ext_vector_type(8)))  float    v8f;

constexpr int    N = 512;
constexpr int    C = 128;                       // C_Z == C_HID == 128
constexpr float  INV_SQRT_N = 0.04419417382415922f; // 1/sqrt(512)
constexpr size_t PLANE = (size_t)N * N;         // 262144

// ---- workspace layout (bytes); requires ~128.125 MB of d_ws ----
constexpr size_t WT_BYTES = (size_t)C * C * sizeof(_Float16); // 32 KB per matrix
constexpr size_t OFF_AWT  = 0;
constexpr size_t OFF_BWT  = WT_BYTES;
constexpr size_t OFF_OWT  = 2 * WT_BYTES;
constexpr size_t OFF_A    = 128 * 1024;                              // a: f16 [c][i][k]
constexpr size_t OFF_B    = OFF_A + (size_t)C * PLANE * sizeof(_Float16); // b: f16 [c][j][k]

DEV_INLINE v8f wmma16x16x32(v16h a, v16h b, v8f c) {
  // (neg_a, A, neg_b, B, c_mod, C, reuse_a, reuse_b)
  return __builtin_amdgcn_wmma_f32_16x16x32_f16(false, a, false, b, (short)0, c,
                                                false, false);
}

// A fragment (16x32 f16): lanes 0-15 row=lane, K=0..15;
// lanes 16-31 row=lane-16, K=16..31. base points at [row0][kc].
DEV_INLINE v16h load_a_frag(const _Float16* base, int stride_h, int lane) {
  const int lm = lane & 15;
  const int ho = (lane >> 4) * 16;
  const v8h lo = *(const v8h*)(base + lm * stride_h + ho);
  const v8h hi = *(const v8h*)(base + lm * stride_h + 8 + ho);
  v16h r;
#pragma unroll
  for (int i = 0; i < 8; ++i) { r[i] = lo[i]; r[8 + i] = hi[i]; }
  return r;
}

// B fragment (32x16 f16): lane holds column lm; K = lane<16 ? 0..15 : 16..31.
// base points at row-major B^T, i.e. base[n*stride + k] = B[k][n] (contiguous in k).
DEV_INLINE v16h load_b_frag(const _Float16* base, int stride_h, int lane) {
  const int lm = lane & 15;
  const int ko = (lane >> 4) * 16;
  return *(const v16h*)(base + lm * stride_h + ko);
}

// ---- CDNA5 async Global->LDS copy (16B per lane), tracked by ASYNCcnt ----
// VDST = VGPR with LDS byte address; VADDR = VGPR pair with 64-bit global addr.
// Generic LDS pointers carry the LDS offset in their low 32 bits (aperture in
// the high 32), so truncation gives a valid LDS address.
DEV_INLINE void async_copy_b128(const _Float16* gsrc, unsigned int lds_off) {
  asm volatile("global_load_async_to_lds_b128 %0, %1, off"
               :: "v"(lds_off), "v"(gsrc)
               : "memory");
}
DEV_INLINE void wait_async0() {
#if __has_builtin(__builtin_amdgcn_s_wait_asynccnt)
  __builtin_amdgcn_s_wait_asynccnt(0);
#else
  asm volatile("s_wait_asynccnt 0x0" ::: "memory");
#endif
}
DEV_INLINE unsigned int lds_off32(const void* p) {
  return (unsigned int)(unsigned long long)p;
}

// ---------------- kernel 0: f32 weights -> f16 transposed ----------------
__global__ __launch_bounds__(256) void k0_prep_weights(
    const float* __restrict__ a_w, const float* __restrict__ b_w,
    const float* __restrict__ out_w,
    _Float16* __restrict__ a_wT, _Float16* __restrict__ b_wT,
    _Float16* __restrict__ out_wT) {
  int t = blockIdx.x * blockDim.x + threadIdx.x;
  if (t < C * C) {
    int k = t / C, n = t % C;
    a_wT[n * C + k]   = (_Float16)a_w[t];
    b_wT[n * C + k]   = (_Float16)b_w[t];
    out_wT[n * C + k] = (_Float16)out_w[t];
  }
}

// ---------------- kernel 1: layernorm + a/b projections ----------------
// block = 256 threads (8 waves), processes 16 rows of the flattened [N*N, C] input.
// Waves 0-3 -> a columns [w*32, w*32+32), waves 4-7 -> b columns.
// Output stored channel-major f16: a_out[c][row], b_out[c][row].
__global__ __launch_bounds__(256) void k1_ln_proj(
    const float* __restrict__ z,
    const float* __restrict__ ln_w, const float* __restrict__ ln_b,
    const float* __restrict__ a_bias, const float* __restrict__ b_bias,
    const _Float16* __restrict__ a_wT, const _Float16* __restrict__ b_wT,
    _Float16* __restrict__ a_out, _Float16* __restrict__ b_out) {
  __shared__ __align__(32) _Float16 xt[16][C];
  __shared__ float red_s[16][16];
  __shared__ float red_q[16][16];
  __shared__ float stat_mu[16];
  __shared__ float stat_rs[16];

  const int t   = threadIdx.x;
  const int row = t >> 4;     // 0..15
  const int seg = t & 15;     // 16 threads per row, 8 floats each
  const size_t row0 = (size_t)blockIdx.x * 16;

  const float* zp = z + (row0 + row) * C + seg * 8;
  float x[8];
#pragma unroll
  for (int i = 0; i < 8; ++i) x[i] = zp[i];
  float s = 0.f, q = 0.f;
#pragma unroll
  for (int i = 0; i < 8; ++i) { s += x[i]; q += x[i] * x[i]; }
  red_s[row][seg] = s;
  red_q[row][seg] = q;
  __syncthreads();
  if (seg == 0) {
    float ts = 0.f, tq = 0.f;
#pragma unroll
    for (int i = 0; i < 16; ++i) { ts += red_s[row][i]; tq += red_q[row][i]; }
    float mu  = ts * (1.0f / C);
    float var = tq * (1.0f / C) - mu * mu;
    stat_mu[row] = mu;
    stat_rs[row] = rsqrtf(var + 1e-5f);
  }
  __syncthreads();
  const float mu = stat_mu[row], rs = stat_rs[row];
#pragma unroll
  for (int i = 0; i < 8; ++i) {
    int cc = seg * 8 + i;
    xt[row][cc] = (_Float16)((x[i] - mu) * rs * ln_w[cc] + ln_b[cc]);
  }
  __syncthreads();

  const int wave = t >> 5;
  const int lane = t & 31;
  const bool is_a = wave < 4;              // wave-uniform
  const int n0 = (is_a ? wave : wave - 4) * 32;
  const _Float16* wT   = is_a ? a_wT : b_wT;
  const float*    bias = is_a ? a_bias : b_bias;
  _Float16*       outp = is_a ? a_out : b_out;

  v8f acc0 = {}, acc1 = {};
#pragma unroll
  for (int kc = 0; kc < C; kc += 32) {
    v16h af  = load_a_frag(&xt[0][kc], C, lane);
    v16h bf0 = load_b_frag(wT + (size_t)n0 * C + kc, C, lane);
    v16h bf1 = load_b_frag(wT + (size_t)(n0 + 16) * C + kc, C, lane);
    acc0 = wmma16x16x32(af, bf0, acc0);
    acc1 = wmma16x16x32(af, bf1, acc1);
  }

  // C/D layout: lane holds column (n0+tile*16+lm), rows rb..rb+7 contiguous.
  const int lm = lane & 15;
  const int rb = (lane >> 4) * 8;
#pragma unroll
  for (int tile = 0; tile < 2; ++tile) {
    v8f  acc = tile ? acc1 : acc0;
    int  col = n0 + tile * 16 + lm;
    float bv = bias[col];
    v8h hv;
#pragma unroll
    for (int v = 0; v < 8; ++v) hv[v] = (_Float16)(acc[v] + bv);
    *(v8h*)(outp + (size_t)col * PLANE + row0 + rb) = hv;  // one 16B store/lane
  }
}

// ---------------- kernel 2: per-channel triangle GEMM U_c = A_c * B_c^T ----------------
// grid (N/64, N/128, C); block = 256 threads (8 waves) computes a 64x128 output
// tile: each wave owns 1 A-frag row-tile and 4 B sub-tiles -> 4 WMMAs per
// K-step between sync points. LDS is staged with async Global->LDS copies
// (ASYNCcnt), double buffered so the copy of K-chunk k+1 overlaps the WMMAs of
// chunk k. Writes u (pre-scaled) as f32 to d_out in [i][j][c] layout.
__global__ __launch_bounds__(256) void k2_triangle(
    const _Float16* __restrict__ a_ws, const _Float16* __restrict__ b_ws,
    float* __restrict__ u) {
  __shared__ __align__(16) _Float16 As[2][64 * 32];   // 2 x 4 KB
  __shared__ __align__(16) _Float16 Bs[2][128 * 32];  // 2 x 8 KB

  const int i0 = blockIdx.x * 64;
  const int j0 = blockIdx.y * 128;
  const int c  = blockIdx.z;
  const int t  = threadIdx.x;
  const int wave = t >> 5, lane = t & 31;
  const int ti  = wave & 3;          // i sub-tile 0..3
  const int tjg = wave >> 2;         // j sub-tile group: {tjg*4 .. tjg*4+3}

  const size_t plane = (size_t)c * PLANE;
  // per-thread 16B copy slots: A = 1 slot, B = 2 slots (rows split 0..63 / 64..127)
  const int frow = t >> 2, fseg = (t & 3) * 8;
  const _Float16* ga  = a_ws + plane + (size_t)(i0 + frow) * N + fseg;
  const _Float16* gb0 = b_ws + plane + (size_t)(j0 + frow) * N + fseg;
  const _Float16* gb1 = b_ws + plane + (size_t)(j0 + 64 + frow) * N + fseg;
  const unsigned int la[2]  = { lds_off32(&As[0][frow * 32 + fseg]),
                                lds_off32(&As[1][frow * 32 + fseg]) };
  const unsigned int lb0[2] = { lds_off32(&Bs[0][frow * 32 + fseg]),
                                lds_off32(&Bs[1][frow * 32 + fseg]) };
  const unsigned int lb1[2] = { lds_off32(&Bs[0][(64 + frow) * 32 + fseg]),
                                lds_off32(&Bs[1][(64 + frow) * 32 + fseg]) };

  // prologue: stage chunk 0 into buffer 0
  async_copy_b128(ga,  la[0]);
  async_copy_b128(gb0, lb0[0]);
  async_copy_b128(gb1, lb1[0]);
  wait_async0();
  __syncthreads();

  v8f acc[4] = {v8f{}, v8f{}, v8f{}, v8f{}};
#pragma unroll
  for (int kc = 0; kc < N; kc += 32) {
    const int cur = (kc >> 5) & 1;
    const int nxt = cur ^ 1;
    if (kc + 32 < N) {               // async-stage next chunk into other buffer
      async_copy_b128(ga  + kc + 32, la[nxt]);
      async_copy_b128(gb0 + kc + 32, lb0[nxt]);
      async_copy_b128(gb1 + kc + 32, lb1[nxt]);
    }

    v16h af = load_a_frag(&As[cur][16 * ti * 32], 32, lane);
#pragma unroll
    for (int tt = 0; tt < 4; ++tt) {
      v16h bf = load_b_frag(&Bs[cur][16 * (tjg * 4 + tt) * 32], 32, lane);
      acc[tt] = wmma16x16x32(af, bf, acc[tt]);
    }

    if (kc + 32 < N) {               // next buffer ready, all waves done reading cur
      wait_async0();
      __syncthreads();
    }
  }

  const int lm = lane & 15;
  const int rb = (lane >> 4) * 8;
#pragma unroll
  for (int tt = 0; tt < 4; ++tt) {
    int j = j0 + (tjg * 4 + tt) * 16 + lm;
#pragma unroll
    for (int v = 0; v < 8; ++v) {
      int i = i0 + ti * 16 + rb + v;
      u[((size_t)i * N + j) * C + c] = acc[tt][v] * INV_SQRT_N;
    }
  }
}

// ---------------- kernel 3: out projection + bias + mask (in-place on d_out) ----------------
// block = 128 threads (4 waves), 16 rows of the flattened [N*N, C] u matrix.
__global__ __launch_bounds__(128) void k3_out_proj(
    const _Float16* __restrict__ out_wT, const float* __restrict__ out_bias,
    const unsigned char* __restrict__ mask,   // residue_mask: numpy bool (1 byte/elem)
    float* __restrict__ u) {
  __shared__ __align__(32) _Float16 xt[16][C];

  const int t = threadIdx.x;
  const size_t row0 = (size_t)blockIdx.x * 16;
  {
    const int row = t >> 3;   // 0..15
    const int seg = t & 7;    // 16 floats each
    const float* up = u + (row0 + row) * C + seg * 16;
#pragma unroll
    for (int i = 0; i < 16; ++i) xt[row][seg * 16 + i] = (_Float16)up[i];
  }
  __syncthreads();

  const int wave = t >> 5, lane = t & 31;
  const int n0 = wave * 32;
  v8f acc0 = {}, acc1 = {};
#pragma unroll
  for (int kc = 0; kc < C; kc += 32) {
    v16h af  = load_a_frag(&xt[0][kc], C, lane);
    v16h bf0 = load_b_frag(out_wT + (size_t)n0 * C + kc, C, lane);
    v16h bf1 = load_b_frag(out_wT + (size_t)(n0 + 16) * C + kc, C, lane);
    acc0 = wmma16x16x32(af, bf0, acc0);
    acc1 = wmma16x16x32(af, bf1, acc1);
  }

  const int lm = lane & 15;
  const int rb = (lane >> 4) * 8;
#pragma unroll
  for (int tile = 0; tile < 2; ++tile) {
    v8f  acc = tile ? acc1 : acc0;
    int  col = n0 + tile * 16 + lm;
    float bv = out_bias[col];
#pragma unroll
    for (int v = 0; v < 8; ++v) {
      size_t r = row0 + rb + v;
      int i = (int)(r / N), j = (int)(r % N);
      float m = (mask[i] && mask[j]) ? 1.0f : 0.0f;
      u[r * C + col] = (acc[v] + bv) * m;
    }
  }
}

// ---------------- host entry ----------------
extern "C" void kernel_launch(void* const* d_in, const int* in_sizes, int n_in,
                              void* d_out, int out_size, void* d_ws, size_t ws_size,
                              hipStream_t stream) {
  (void)in_sizes; (void)n_in; (void)out_size; (void)ws_size;
  const float*         z     = (const float*)d_in[0];
  const unsigned char* mask  = (const unsigned char*)d_in[1];
  const float*         ln_w  = (const float*)d_in[2];
  const float*         ln_b  = (const float*)d_in[3];
  const float*         a_w   = (const float*)d_in[4];
  const float*         a_b   = (const float*)d_in[5];
  const float*         b_w   = (const float*)d_in[6];
  const float*         b_b   = (const float*)d_in[7];
  const float*         out_w = (const float*)d_in[8];
  const float*         out_b = (const float*)d_in[9];
  float* out = (float*)d_out;
  char*  ws  = (char*)d_ws;

  _Float16* a_wT   = (_Float16*)(ws + OFF_AWT);
  _Float16* b_wT   = (_Float16*)(ws + OFF_BWT);
  _Float16* out_wT = (_Float16*)(ws + OFF_OWT);
  _Float16* a_h    = (_Float16*)(ws + OFF_A);
  _Float16* b_h    = (_Float16*)(ws + OFF_B);

  k0_prep_weights<<<(C * C + 255) / 256, 256, 0, stream>>>(a_w, b_w, out_w,
                                                           a_wT, b_wT, out_wT);
  k1_ln_proj<<<(N * N) / 16, 256, 0, stream>>>(z, ln_w, ln_b, a_b, b_b,
                                               a_wT, b_wT, a_h, b_h);
  k2_triangle<<<dim3(N / 64, N / 128, C), 256, 0, stream>>>(a_h, b_h, out);
  k3_out_proj<<<(N * N) / 16, 128, 0, stream>>>(out_wT, out_b, mask, out);
}
